// S4Block_63247688401497
// MI455X (gfx1250) — matrix operand order repaired and checked by reading, not compiled, not run on hardware
//
#include <hip/hip_runtime.h>
#include <hip/hip_bf16.h>
#include <cstdint>

// S4 block, B=16, L=2048, H=512, N=32, DEPTH=2.
// Pipeline per layer:
//   1) prep:    w = exp(dt*A), C' = C*(w-1)/A, w^Lc   (transposed layouts for coalesced scan loads)
//   2) conv_wo: Wo -> f16
//   3) scan p1: chunk-local state scans (zero init) -> carry-out per chunk
//   4) scan p2: sequential prefix-combine of carries with w^Lc (per b,h,n)
//   5) scan p3: rescan with carry-in, y = 2*Re(C'·s) + D*u, GELU, store f16 (GEMM A operand)
//   6) gemm:    z = y @ Wo^T + bo via v_wmma_f32_16x16x32_f16, fused GLU epilogue.
//               Wave tile = 32M x (32 + 32-gate)N: 8 WMMAs per 12 b128 loads.

namespace {
constexpr int S4B  = 16;
constexpr int S4L  = 2048;
constexpr int S4H  = 512;
constexpr int S4N  = 32;
constexpr int S4CH = 16;              // chunks along L
constexpr int S4LC = S4L / S4CH;      // 128
constexpr int S4F  = 2 * S4H;         // 1024
}

typedef __attribute__((ext_vector_type(16))) _Float16 v16h;
typedef __attribute__((ext_vector_type(8)))  _Float16 v8h;
typedef __attribute__((ext_vector_type(8)))  float    v8f;

union V16U { v16h v; v8h h[2]; };

__device__ __forceinline__ v16h load_frag16(const _Float16* p0, const _Float16* p1) {
  V16U u;
  u.h[0] = *reinterpret_cast<const v8h*>(p0);
  u.h[1] = *reinterpret_cast<const v8h*>(p1);
  return u.v;
}

__device__ __forceinline__ float gelu_tanh(float x) {
  float x3 = x * x * x;
  float t  = tanhf(0.7978845608028654f * (x + 0.044715f * x3));
  return 0.5f * x * (1.f + t);
}

__device__ __forceinline__ float sigmoidf_(float x) {
  return 1.f / (1.f + __expf(-x));
}

// ---------------------------------------------------------------- prep ------
__global__ void __launch_bounds__(256) s4_prep(
    const float* __restrict__ log_dt, const float* __restrict__ A_re,
    const float* __restrict__ A_im,  const float* __restrict__ C_re,
    const float* __restrict__ C_im,
    float* __restrict__ wre_t, float* __restrict__ wim_t,
    float* __restrict__ cre_t, float* __restrict__ cim_t,
    float* __restrict__ wLc_re, float* __restrict__ wLc_im) {
  int idx = blockIdx.x * blockDim.x + threadIdx.x;
  if (idx >= S4H * S4N) return;
  int h = idx >> 5;
  int n = idx & (S4N - 1);
  float dt  = __expf(log_dt[h]);
  float ar  = A_re[idx], ai = A_im[idx];
  float dre = -dt * ar,  dim = dt * ai;     // dt*A, A = -ar + i*ai
  float e   = __expf(dre);
  float wr  = e * __cosf(dim), wi = e * __sinf(dim);   // w = exp(dtA)
  // f = (w - 1) / A
  float nr = wr - 1.f, ni = wi;
  float denr = -ar, deni = ai;
  float inv = 1.f / (denr * denr + deni * deni);
  float fr = (nr * denr + ni * deni) * inv;
  float fi = (ni * denr - nr * deni) * inv;
  float cr = C_re[idx], ci = C_im[idx];
  wre_t[n * S4H + h] = wr;
  wim_t[n * S4H + h] = wi;
  cre_t[n * S4H + h] = cr * fr - ci * fi;   // C' = C * f
  cim_t[n * S4H + h] = cr * fi + ci * fr;
  // w^Lc for chunk carry combination
  float eL  = __expf(dre * (float)S4LC);
  float phL = dim * (float)S4LC;
  wLc_re[idx] = eL * __cosf(phL);
  wLc_im[idx] = eL * __sinf(phL);
}

// ---------------------------------------------------------------- Wo->f16 ---
__global__ void __launch_bounds__(256) s4_conv_wo(
    const float* __restrict__ Wo, _Float16* __restrict__ wo16) {
  int idx = blockIdx.x * blockDim.x + threadIdx.x;
  if (idx >= S4F * S4H) return;
  wo16[idx] = (_Float16)Wo[idx];
}

// ------------------------------------------------------------ scan phase 1 --
// thread = (chunk, b, h), h fastest -> coalesced u loads. Zero-init local scan,
// emit carry-out state (N complex values).
__global__ void __launch_bounds__(256) s4_scan_phase1(
    const float* __restrict__ uin, const int* __restrict__ lengths,
    const float* __restrict__ wre_t, const float* __restrict__ wim_t,
    float* __restrict__ carry_re, float* __restrict__ carry_im) {
  int tid = blockIdx.x * blockDim.x + threadIdx.x;
  if (tid >= S4B * S4H * S4CH) return;
  int h     = tid & (S4H - 1);
  int rest  = tid >> 9;
  int b     = rest & (S4B - 1);
  int chunk = rest >> 4;
  int len   = lengths[b];

  float wr[S4N], wi[S4N], sr[S4N], si[S4N];
#pragma unroll
  for (int n = 0; n < S4N; ++n) {
    wr[n] = wre_t[n * S4H + h];
    wi[n] = wim_t[n * S4H + h];
    sr[n] = 0.f; si[n] = 0.f;
  }
  int l0 = chunk * S4LC;
  const float* up = uin + ((size_t)b * S4L + l0) * S4H + h;
  for (int i = 0; i < S4LC; ++i) {
    int   l  = l0 + i;
    float um = (l < len) ? up[(size_t)i * S4H] : 0.f;
#pragma unroll
    for (int n = 0; n < S4N; ++n) {
      float nr = fmaf(wr[n], sr[n], fmaf(-wi[n], si[n], um));
      float ni = fmaf(wr[n], si[n], wi[n] * sr[n]);
      sr[n] = nr; si[n] = ni;
    }
  }
  size_t cidx = (((size_t)chunk * S4B + b) * S4H + h) * S4N;
#pragma unroll
  for (int n = 0; n < S4N; ++n) {
    carry_re[cidx + n] = sr[n];
    carry_im[cidx + n] = si[n];
  }
}

// ------------------------------------------------------------ scan phase 2 --
// thread = (b,h,n): sequential prefix over 16 chunks; replaces carry-out with
// carry-IN (state entering each chunk).
__global__ void __launch_bounds__(256) s4_scan_phase2(
    const float* __restrict__ wLc_re, const float* __restrict__ wLc_im,
    float* __restrict__ carry_re, float* __restrict__ carry_im) {
  int tid = blockIdx.x * blockDim.x + threadIdx.x;
  if (tid >= S4B * S4H * S4N) return;
  int n  = tid & (S4N - 1);
  int bh = tid >> 5;
  int h  = bh & (S4H - 1);
  float wr = wLc_re[h * S4N + n];
  float wi = wLc_im[h * S4N + n];
  float cr = 0.f, ci = 0.f;
  size_t stride = (size_t)S4B * S4H * S4N;
  size_t base   = (size_t)tid;   // (b*H + h)*N + n
  for (int c = 0; c < S4CH; ++c) {
    size_t idx = (size_t)c * stride + base;
    float tr = carry_re[idx], ti = carry_im[idx];
    carry_re[idx] = cr; carry_im[idx] = ci;
    float ncr = fmaf(wr, cr, fmaf(-wi, ci, tr));
    float nci = fmaf(wr, ci, fmaf(wi, cr, ti));
    cr = ncr; ci = nci;
  }
}

// ------------------------------------------------------------ scan phase 3 --
// Rescan with carry-in; y = 2*Re(C'·s) + D*u; GELU; emit f16 GEMM A operand.
__global__ void __launch_bounds__(256) s4_scan_phase3(
    const float* __restrict__ uin, const int* __restrict__ lengths,
    const float* __restrict__ wre_t, const float* __restrict__ wim_t,
    const float* __restrict__ cre_t, const float* __restrict__ cim_t,
    const float* __restrict__ carry_re, const float* __restrict__ carry_im,
    const float* __restrict__ Dp, _Float16* __restrict__ y16) {
  int tid = blockIdx.x * blockDim.x + threadIdx.x;
  if (tid >= S4B * S4H * S4CH) return;
  int h     = tid & (S4H - 1);
  int rest  = tid >> 9;
  int b     = rest & (S4B - 1);
  int chunk = rest >> 4;
  int len   = lengths[b];
  float dh  = Dp[h];

  float wr[S4N], wi[S4N], pr[S4N], pi[S4N], sr[S4N], si[S4N];
  size_t cidx = (((size_t)chunk * S4B + b) * S4H + h) * S4N;
#pragma unroll
  for (int n = 0; n < S4N; ++n) {
    wr[n] = wre_t[n * S4H + h];
    wi[n] = wim_t[n * S4H + h];
    pr[n] = cre_t[n * S4H + h];
    pi[n] = cim_t[n * S4H + h];
    sr[n] = carry_re[cidx + n];
    si[n] = carry_im[cidx + n];
  }
  int l0 = chunk * S4LC;
  const float* up = uin + ((size_t)b * S4L + l0) * S4H + h;
  _Float16*    yp = y16 + ((size_t)b * S4L + l0) * S4H + h;
  for (int i = 0; i < S4LC; ++i) {
    int   l  = l0 + i;
    float um = (l < len) ? up[(size_t)i * S4H] : 0.f;
    float acc = 0.f;
#pragma unroll
    for (int n = 0; n < S4N; ++n) {
      float nr = fmaf(wr[n], sr[n], fmaf(-wi[n], si[n], um));
      float ni = fmaf(wr[n], si[n], wi[n] * sr[n]);
      sr[n] = nr; si[n] = ni;
      acc = fmaf(pr[n], sr[n], fmaf(-pi[n], si[n], acc));
    }
    float y = fmaf(dh, um, 2.f * acc);
    yp[(size_t)i * S4H] = (_Float16)gelu_tanh(y);
  }
}

// ----------------------------------------------------------------- GEMM -----
// z = y16 @ Wo^T + bo, fused GLU: out[m,f] = z[m,f]*sigmoid(z[m,f+512]).
// Wave tile: 32 rows (2 M-frags) x (32 cols + their GLU partners) -> 8 accs.
// B fragments are reused across both M-frags: 8 WMMAs per 12 b128 loads.
// A frag (16x32 f16): lane<16 -> row=lane, K=0..15; lane>=16 -> row=lane-16, K=16..31.
// B frag (32x16 f16): lane<16 -> col=lane, K=0..15; lane>=16 -> col=lane-16, K=16..31.
__global__ void __launch_bounds__(256) s4_gemm_glu(
    const _Float16* __restrict__ y16, const _Float16* __restrict__ wo16,
    const float* __restrict__ bo, float* __restrict__ outp) {
  int wave = threadIdx.x >> 5;
  int lane = threadIdx.x & 31;
  int m0 = blockIdx.x * 256 + wave * 32;   // 32 rows per wave, 8 waves -> 256/block
  int f0 = blockIdx.y * 32;

  int rc     = lane & 15;          // A row / B col within tile
  int koff_a = (lane >> 4) * 16;   // A K-half select
  int koff_b = (lane >> 4) * 16;   // B K-half select

  const _Float16* aBase0 = y16 + ((size_t)(m0 + rc)) * S4H + koff_a;
  const _Float16* aBase1 = aBase0 + (size_t)16 * S4H;
  const _Float16* b0Base = wo16 + ((size_t)(f0 + rc)) * S4H + koff_b;          // cols f0..f0+15
  const _Float16* b1Base = b0Base + (size_t)16 * S4H;                          // cols f0+16..f0+31
  const _Float16* b2Base = wo16 + ((size_t)(S4H + f0 + rc)) * S4H + koff_b;    // GLU gate half
  const _Float16* b3Base = b2Base + (size_t)16 * S4H;

  v8f acc00 = {}, acc01 = {}, acc02 = {}, acc03 = {};
  v8f acc10 = {}, acc11 = {}, acc12 = {}, acc13 = {};
  for (int k = 0; k < S4H; k += 32) {
    __builtin_prefetch(aBase0 + k + 32, 0, 3);   // global_prefetch_b8
    __builtin_prefetch(aBase1 + k + 32, 0, 3);
    v16h a0 = load_frag16(aBase0 + k, aBase0 + k + 8);
    v16h a1 = load_frag16(aBase1 + k, aBase1 + k + 8);
    v16h b0 = load_frag16(b0Base + k, b0Base + k + 8);
    v16h b1 = load_frag16(b1Base + k, b1Base + k + 8);
    v16h b2 = load_frag16(b2Base + k, b2Base + k + 8);
    v16h b3 = load_frag16(b3Base + k, b3Base + k + 8);
    acc00 = __builtin_amdgcn_wmma_f32_16x16x32_f16(false, a0, false, b0, (short)0, acc00, false, false);
    acc01 = __builtin_amdgcn_wmma_f32_16x16x32_f16(false, a0, false, b1, (short)0, acc01, false, false);
    acc02 = __builtin_amdgcn_wmma_f32_16x16x32_f16(false, a0, false, b2, (short)0, acc02, false, false);
    acc03 = __builtin_amdgcn_wmma_f32_16x16x32_f16(false, a0, false, b3, (short)0, acc03, false, false);
    acc10 = __builtin_amdgcn_wmma_f32_16x16x32_f16(false, a1, false, b0, (short)0, acc10, false, false);
    acc11 = __builtin_amdgcn_wmma_f32_16x16x32_f16(false, a1, false, b1, (short)0, acc11, false, false);
    acc12 = __builtin_amdgcn_wmma_f32_16x16x32_f16(false, a1, false, b2, (short)0, acc12, false, false);
    acc13 = __builtin_amdgcn_wmma_f32_16x16x32_f16(false, a1, false, b3, (short)0, acc13, false, false);
  }

  // C/D layout: vgpr r -> row r (+8 for lanes>=16), col = lane&15
  int rowOff = (lane >> 4) * 8;
  float bo0 = bo[f0 + rc];
  float bo1 = bo[f0 + 16 + rc];
  float bo2 = bo[S4H + f0 + rc];
  float bo3 = bo[S4H + f0 + 16 + rc];
#pragma unroll
  for (int r = 0; r < 8; ++r) {
    size_t row0 = (size_t)(m0 + r + rowOff);
    size_t row1 = row0 + 16;
    float z1, z2;
    z1 = acc00[r] + bo0; z2 = acc02[r] + bo2;
    outp[row0 * S4H + f0 + rc]      = z1 * sigmoidf_(z2);
    z1 = acc01[r] + bo1; z2 = acc03[r] + bo3;
    outp[row0 * S4H + f0 + 16 + rc] = z1 * sigmoidf_(z2);
    z1 = acc10[r] + bo0; z2 = acc12[r] + bo2;
    outp[row1 * S4H + f0 + rc]      = z1 * sigmoidf_(z2);
    z1 = acc11[r] + bo1; z2 = acc13[r] + bo3;
    outp[row1 * S4H + f0 + 16 + rc] = z1 * sigmoidf_(z2);
  }
}

// ---------------------------------------------------------------- launch ----
extern "C" void kernel_launch(void* const* d_in, const int* in_sizes, int n_in,
                              void* d_out, int out_size, void* d_ws, size_t ws_size,
                              hipStream_t stream) {
  (void)in_sizes; (void)n_in; (void)out_size; (void)ws_size;
  const float* u       = (const float*)d_in[0];
  const int*   lengths = (const int*)  d_in[1];
  const float* log_dt  = (const float*)d_in[2];
  const float* A_re    = (const float*)d_in[3];
  const float* A_im    = (const float*)d_in[4];
  const float* C_re    = (const float*)d_in[5];
  const float* C_im    = (const float*)d_in[6];
  const float* Dp      = (const float*)d_in[7];
  const float* Wo      = (const float*)d_in[8];
  const float* bo      = (const float*)d_in[9];

  char* ws = (char*)d_ws;
  size_t off = 0;
  auto take = [&](size_t bytes) -> char* {
    char* p = ws + off;
    off = (off + bytes + 255) & ~(size_t)255;
    return p;
  };
  const size_t HN = (size_t)S4H * S4N;
  float*    wre_t   = (float*)take(HN * 4);
  float*    wim_t   = (float*)take(HN * 4);
  float*    cre_t   = (float*)take(HN * 4);
  float*    cim_t   = (float*)take(HN * 4);
  float*    wLc_re  = (float*)take(HN * 4);
  float*    wLc_im  = (float*)take(HN * 4);
  _Float16* wo16    = (_Float16*)take((size_t)S4F * S4H * 2);
  float*    carryRe = (float*)take((size_t)S4CH * S4B * S4H * S4N * 4);
  float*    carryIm = (float*)take((size_t)S4CH * S4B * S4H * S4N * 4);
  _Float16* y16     = (_Float16*)take((size_t)S4B * S4L * S4H * 2);
  float*    act     = (float*)take((size_t)S4B * S4L * S4H * 4);

  const int prepBlocks  = (S4H * S4N) / 256;          // 64
  const int convBlocks  = (S4F * S4H) / 256;          // 2048
  const int p1Blocks    = (S4B * S4H * S4CH) / 256;   // 512
  const int p2Blocks    = (S4B * S4H * S4N) / 256;    // 1024
  const dim3 gemmGrid((S4B * S4L) / 256, S4H / 32);   // (128, 16)

  for (int layer = 0; layer < 2; ++layer) {
    const float* uin  = (layer == 0) ? u : act;
    float*       outp = (layer == 1) ? (float*)d_out : act;
    const size_t pL   = (size_t)layer;

    s4_prep<<<prepBlocks, 256, 0, stream>>>(
        log_dt + pL * S4H, A_re + pL * HN, A_im + pL * HN,
        C_re + pL * HN, C_im + pL * HN,
        wre_t, wim_t, cre_t, cim_t, wLc_re, wLc_im);

    s4_conv_wo<<<convBlocks, 256, 0, stream>>>(
        Wo + pL * (size_t)S4F * S4H, wo16);

    s4_scan_phase1<<<p1Blocks, 256, 0, stream>>>(
        uin, lengths, wre_t, wim_t, carryRe, carryIm);

    s4_scan_phase2<<<p2Blocks, 256, 0, stream>>>(
        wLc_re, wLc_im, carryRe, carryIm);

    s4_scan_phase3<<<p1Blocks, 256, 0, stream>>>(
        uin, lengths, wre_t, wim_t, cre_t, cim_t,
        carryRe, carryIm, Dp + pL * S4H, y16);

    s4_gemm_glu<<<gemmGrid, 256, 0, stream>>>(
        y16, wo16, bo + pL * S4F, outp);
  }
}
